// Transformer_12369505813170
// MI455X (gfx1250) — compile-verified, matrix operand order checked by reading, not run on hardware
//
#include <hip/hip_runtime.h>
#include <hip/hip_bf16.h>

typedef _Float16 half_t;
typedef __attribute__((ext_vector_type(16))) _Float16 v16h;
typedef __attribute__((ext_vector_type(8)))  _Float16 v8h;
typedef __attribute__((ext_vector_type(8)))  float    v8f;

#define SEQ     4096
#define DMODEL  1024
#define DFF     4096
#define NHEAD   8
#define DKH     128

// ---------------------------------------------------------------------------
// CDNA5 async global->LDS copy (GLOBAL_LOAD_ASYNC_TO_LDS_B128, ASYNCcnt).
// LDS destination VGPR holds the LDS byte address = low 32 bits of the flat
// shared-space pointer (aperture keeps the LDS offset in ADDR[31:0]).
// Async loads retire in order, so waiting ASYNCcnt <= N releases the oldest
// outstanding copies first -> enables double-buffered pipelining.
// ---------------------------------------------------------------------------
__device__ __forceinline__ void async_copy_b128(const half_t* gptr, void* lptr) {
    const unsigned lds = (unsigned)(uintptr_t)lptr;
    const unsigned long long ga = (unsigned long long)(uintptr_t)gptr;
    asm volatile("global_load_async_to_lds_b128 %0, %1, off"
                 :: "v"(lds), "v"(ga) : "memory");
}
__device__ __forceinline__ void wait_async_le0() {
    asm volatile("s_wait_asynccnt 0x0" ::: "memory");
}
__device__ __forceinline__ void wait_async_le4() {
    asm volatile("s_wait_asynccnt 0x4" ::: "memory");
}
__device__ __forceinline__ void wait_async_le8() {
    asm volatile("s_wait_asynccnt 0x8" ::: "memory");
}

// ---------------------------------------------------------------------------
// Conversion / packing (once per launch, ~48MB, bandwidth-trivial).
// All weights stored TRANSPOSED [N][K] f16 so WMMA B-fragments load as b128.
// ---------------------------------------------------------------------------
__global__ void cvt_f32_f16(const float* __restrict__ in, half_t* __restrict__ out, int n) {
    int i = blockIdx.x * blockDim.x + threadIdx.x;
    if (i < n) out[i] = (half_t)in[i];
}

__global__ void cvt_t_f32_f16(const float* __restrict__ in, half_t* __restrict__ out,
                              int K, int N) {
    int i = blockIdx.x * blockDim.x + threadIdx.x;
    if (i >= K * N) return;
    int n = i / K;
    int k = i % K;
    out[i] = (half_t)in[(size_t)k * N + n];
}

__global__ void pack_qk_wt(const float* __restrict__ WQ, const float* __restrict__ WK,
                           half_t* __restrict__ out) {
    int i = blockIdx.x * blockDim.x + threadIdx.x;
    if (i >= 2 * DMODEL * DMODEL) return;
    int n = i >> 10;
    int d = i & 1023;
    int which = n >> 10;
    int h = (n & 1023) >> 7;
    int k = n & 127;
    const float* W = which ? WK : WQ;
    out[i] = (half_t)W[(size_t)h * DMODEL * DKH + (size_t)d * DKH + k];
}

__global__ void pack_v_wt(const float* __restrict__ WV, half_t* __restrict__ out) {
    int i = blockIdx.x * blockDim.x + threadIdx.x;
    if (i >= DMODEL * DMODEL) return;
    int n = i >> 10;
    int d = i & 1023;
    int h = n >> 7;
    int k = n & 127;
    out[i] = (half_t)WV[(size_t)h * DMODEL * DKH + (size_t)d * DKH + k];
}

__global__ void concat2_f32(const float* __restrict__ a, const float* __restrict__ b,
                            float* __restrict__ out, int n) {
    int i = blockIdx.x * blockDim.x + threadIdx.x;
    if (i < 2 * n) out[i] = (i < n) ? a[i] : b[i - n];
}

// ---------------------------------------------------------------------------
// WMMA GEMM: C[M,N] = A[M,K](f16) * BT[N,K](f16)^T (+bias)(+resid f32)
// 256 threads = 8 waves; block tile 128x128; each wave 32x64 (2x4 frags,
// 8 WMMAs per 32-k step). Double-buffered async global->LDS staging.
// ---------------------------------------------------------------------------
#define GBM 128
#define GBN 128
#define GBK 32

__global__ __launch_bounds__(256)
void gemm_wmma(const half_t* __restrict__ A, const half_t* __restrict__ BT,
               const float* __restrict__ bias, const float* __restrict__ resid,
               half_t* __restrict__ outH, float* __restrict__ outF,
               half_t* __restrict__ outHT,   // transposed f16 output [N][M]
               int M, int N, int K, int relu) {
    __shared__ __align__(16) half_t As[2][GBM][40];   // [buf][m][k]
    __shared__ __align__(16) half_t Bs[2][GBN][40];   // [buf][n][k]

    const int tid  = threadIdx.x;
    const int lane = tid & 31;
    const int wave = tid >> 5;
    const int g    = lane >> 4;
    const int li   = lane & 15;
    const int wm   = (wave >> 1) << 5;   // 0,32,64,96
    const int wn   = (wave & 1) << 6;    // 0,64
    const int bm   = blockIdx.y * GBM;
    const int bn   = blockIdx.x * GBN;
    const int sr   = tid >> 2;           // 0..63
    const int sc   = (tid & 3) << 3;     // 0,8,16,24

    auto issue_stage = [&](int kk, int buf) {
        async_copy_b128(A  + (size_t)(bm + sr)      * K + kk + sc, &As[buf][sr][sc]);
        async_copy_b128(A  + (size_t)(bm + sr + 64) * K + kk + sc, &As[buf][sr + 64][sc]);
        async_copy_b128(BT + (size_t)(bn + sr)      * K + kk + sc, &Bs[buf][sr][sc]);
        async_copy_b128(BT + (size_t)(bn + sr + 64) * K + kk + sc, &Bs[buf][sr + 64][sc]);
    };

    v8f acc[2][4] = {};
    const int nk = K / GBK;
    issue_stage(0, 0);

    for (int it = 0; it < nk; ++it) {
        const int cur = it & 1;
        if (it + 1 < nk) {
            issue_stage((it + 1) * GBK, cur ^ 1);   // prefetch next stage
            wait_async_le4();                        // stage `it` fully in LDS
        } else {
            wait_async_le0();
        }
        __syncthreads();

        v16h af[2], bf[4];
#pragma unroll
        for (int i = 0; i < 2; ++i) {
            union { uint4 q[2]; v16h v; } ua;
            ua.q[0] = *(const uint4*)&As[cur][wm + i * 16 + li][(g << 3)];
            ua.q[1] = *(const uint4*)&As[cur][wm + i * 16 + li][16 + (g << 3)];
            af[i] = ua.v;
        }
#pragma unroll
        for (int j = 0; j < 4; ++j) {
            union { uint4 q[2]; v16h v; } ub;
            ub.q[0] = *(const uint4*)&Bs[cur][wn + j * 16 + li][(g << 3)];
            ub.q[1] = *(const uint4*)&Bs[cur][wn + j * 16 + li][16 + (g << 3)];
            bf[j] = ub.v;
        }
#pragma unroll
        for (int i = 0; i < 2; ++i)
#pragma unroll
            for (int j = 0; j < 4; ++j)
                acc[i][j] = __builtin_amdgcn_wmma_f32_16x16x32_f16(
                    false, af[i], false, bf[j], (short)0, acc[i][j], false, false);
        __syncthreads();   // protects the buffer the next issue will overwrite
    }

#pragma unroll
    for (int i = 0; i < 2; ++i) {
#pragma unroll
        for (int j = 0; j < 4; ++j) {
            const int col  = bn + wn + j * 16 + li;
            const int rowb = bm + wm + i * 16 + (g << 3);
            const float bc = bias ? bias[col] : 0.0f;
            if (outHT) {
                v8h hv;
#pragma unroll
                for (int r = 0; r < 8; ++r) {
                    float v = acc[i][j][r] + bc;
                    if (relu) v = fmaxf(v, 0.0f);
                    hv[r] = (half_t)v;
                }
                *(v8h*)(outHT + (size_t)col * M + rowb) = hv;
            } else {
#pragma unroll
                for (int r = 0; r < 8; ++r) {
                    const int row = rowb + r;
                    float v = acc[i][j][r] + bc;
                    if (resid) v += resid[(size_t)row * N + col];
                    if (relu)  v = fmaxf(v, 0.0f);
                    if (outF)  outF[(size_t)row * N + col] = v;
                    if (outH)  outH[(size_t)row * N + col] = (half_t)v;
                }
            }
        }
    }
}

// ---------------------------------------------------------------------------
// Flash-style attention. 128 threads (4 waves) = 64 queries of one head;
// 32-key K/Vt tiles double-buffered, async-staged, shared by all waves.
// QK: [SEQ, 2048] f16 (Q cols 0..1023, K cols 1024..2047).
// VT: [1024][SEQ] f16 (row = h*128+dk). ATT: [SEQ,1024] f16.
// ---------------------------------------------------------------------------
__global__ __launch_bounds__(128)
void attention_wmma(const half_t* __restrict__ QK, const half_t* __restrict__ VT,
                    half_t* __restrict__ ATT) {
    __shared__ __align__(16) half_t Ks[2][32][136];   // [buf][key][dk]
    __shared__ __align__(16) half_t Vs[2][128][40];   // [buf][dk][key]
    __shared__ __align__(16) half_t Ps[4][16][32];    // per-wave P relayout

    const int tid   = threadIdx.x;
    const int lane  = tid & 31;
    const int wave  = tid >> 5;
    const int g     = lane >> 4;
    const int li    = lane & 15;
    const int h     = blockIdx.y;
    const int qbase = (blockIdx.x << 6) + (wave << 4);
    const int ldq   = 2 * DMODEL;
    const int qc    = h * DKH;
    const int kc    = DMODEL + h * DKH;

    auto issue_kv = [&](int kb2, int buf) {
#pragma unroll
        for (int itr = 0; itr < 4; ++itr) {
            const int s    = tid + (itr << 7);     // 0..511
            const int kr   = s >> 4;               // 0..31
            const int kcol = (s & 15) << 3;        // 0..120
            async_copy_b128(QK + (size_t)(kb2 + kr) * ldq + kc + kcol, &Ks[buf][kr][kcol]);
            const int vr   = s >> 2;               // 0..127
            const int vcol = (s & 3) << 3;         // 0,8,16,24
            async_copy_b128(VT + (size_t)(h * DKH + vr) * SEQ + kb2 + vcol, &Vs[buf][vr][vcol]);
        }
    };

    v16h afQ[4];
#pragma unroll
    for (int kb = 0; kb < 4; ++kb) {
        union { uint4 q[2]; v16h v; } u;
        const half_t* p = QK + (size_t)(qbase + li) * ldq + qc + (kb << 5) + (g << 3);
        u.q[0] = *(const uint4*)p;
        u.q[1] = *(const uint4*)(p + 16);
        afQ[kb] = u.v;
    }

    float m[8], l[8];
    v8f o[8];
    const v8f vzero = {};
#pragma unroll
    for (int r = 0; r < 8; ++r) { m[r] = -1e30f; l[r] = 0.0f; }
#pragma unroll
    for (int ct = 0; ct < 8; ++ct) o[ct] = vzero;

    const int nt = SEQ / 32;
    issue_kv(0, 0);

    for (int t = 0; t < nt; ++t) {
        const int cur = t & 1;
        if (t + 1 < nt) {
            issue_kv((t + 1) * 32, cur ^ 1);
            wait_async_le8();
        } else {
            wait_async_le0();
        }
        __syncthreads();

        // ---- scores S = Q * K^T for 32 keys ----
        v8f s0 = vzero, s1 = vzero;
#pragma unroll
        for (int kb = 0; kb < 4; ++kb) {
            union { uint4 q[2]; v16h v; } u0, u1;
            u0.q[0] = *(const uint4*)&Ks[cur][li][(kb << 5) + (g << 3)];
            u0.q[1] = *(const uint4*)&Ks[cur][li][(kb << 5) + 16 + (g << 3)];
            u1.q[0] = *(const uint4*)&Ks[cur][16 + li][(kb << 5) + (g << 3)];
            u1.q[1] = *(const uint4*)&Ks[cur][16 + li][(kb << 5) + 16 + (g << 3)];
            s0 = __builtin_amdgcn_wmma_f32_16x16x32_f16(false, afQ[kb], false, u0.v,
                                                        (short)0, s0, false, false);
            s1 = __builtin_amdgcn_wmma_f32_16x16x32_f16(false, afQ[kb], false, u1.v,
                                                        (short)0, s1, false, false);
        }
        s0 = s0 * 0.125f;
        s1 = s1 * 0.125f;

        // ---- online softmax (rows g*8+r live in 16-lane halves) ----
#pragma unroll
        for (int r = 0; r < 8; ++r) {
            float mx = fmaxf(s0[r], s1[r]);
#pragma unroll
            for (int off = 8; off > 0; off >>= 1) mx = fmaxf(mx, __shfl_xor(mx, off, 16));
            const float mn = fmaxf(m[r], mx);
            const float sc = __expf(m[r] - mn);
            const float p0 = __expf(s0[r] - mn);
            const float p1 = __expf(s1[r] - mn);
            float sum = p0 + p1;
#pragma unroll
            for (int off = 8; off > 0; off >>= 1) sum += __shfl_xor(sum, off, 16);
            l[r] = l[r] * sc + sum;
            m[r] = mn;
#pragma unroll
            for (int ct = 0; ct < 8; ++ct) o[ct][r] = o[ct][r] * sc;
            Ps[wave][(g << 3) + r][li]      = (half_t)p0;
            Ps[wave][(g << 3) + r][16 + li] = (half_t)p1;
        }
        asm volatile("" ::: "memory");   // wave-internal LDS relayout ordering
        union { uint4 q[2]; v16h v; } up;
        up.q[0] = *(const uint4*)&Ps[wave][li][(g << 3)];
        up.q[1] = *(const uint4*)&Ps[wave][li][16 + (g << 3)];
        const v16h ap = up.v;
        asm volatile("" ::: "memory");

        // ---- O += P * V ----
#pragma unroll
        for (int ct = 0; ct < 8; ++ct) {
            union { uint4 q[2]; v16h v; } uv;
            uv.q[0] = *(const uint4*)&Vs[cur][(ct << 4) + li][(g << 3)];
            uv.q[1] = *(const uint4*)&Vs[cur][(ct << 4) + li][16 + (g << 3)];
            o[ct] = __builtin_amdgcn_wmma_f32_16x16x32_f16(false, ap, false, uv.v,
                                                           (short)0, o[ct], false, false);
        }
        __syncthreads();
    }

#pragma unroll
    for (int ct = 0; ct < 8; ++ct) {
#pragma unroll
        for (int r = 0; r < 8; ++r) {
            const int row = qbase + (g << 3) + r;
            const int col = h * DKH + (ct << 4) + li;
            ATT[(size_t)row * DMODEL + col] = (half_t)(o[ct][r] / l[r]);
        }
    }
}

// ---------------------------------------------------------------------------
// Row LayerNorm: y = g*(x-mean)/(std+eps)+b, std unbiased (ddof=1).
// ---------------------------------------------------------------------------
__global__ void layernorm_row(const float* __restrict__ X, const float* __restrict__ gam,
                              const float* __restrict__ bet, float* __restrict__ outF,
                              half_t* __restrict__ outH, int N) {
    __shared__ float s_sum[256];
    __shared__ float s_sq[256];
    const int row = blockIdx.x;
    const int tid = threadIdx.x;

    float s = 0.0f, s2 = 0.0f;
    for (int c = tid; c < N; c += 256) {
        const float v = X[(size_t)row * N + c];
        s += v; s2 += v * v;
    }
    s_sum[tid] = s; s_sq[tid] = s2;
    __syncthreads();
    for (int st = 128; st > 0; st >>= 1) {
        if (tid < st) { s_sum[tid] += s_sum[tid + st]; s_sq[tid] += s_sq[tid + st]; }
        __syncthreads();
    }
    const float mean = s_sum[0] / (float)N;
    float var = (s_sq[0] - (float)N * mean * mean) / (float)(N - 1);
    var = fmaxf(var, 0.0f);
    const float inv = 1.0f / (sqrtf(var) + 1e-6f);

    for (int c = tid; c < N; c += 256) {
        const float v = (X[(size_t)row * N + c] - mean) * inv * gam[c] + bet[c];
        if (outF) outF[(size_t)row * N + c] = v;
        if (outH) outH[(size_t)row * N + c] = (half_t)v;
    }
}

// ---------------------------------------------------------------------------
// Orchestration
// ---------------------------------------------------------------------------
extern "C" void kernel_launch(void* const* d_in, const int* in_sizes, int n_in,
                              void* d_out, int out_size, void* d_ws, size_t ws_size,
                              hipStream_t stream) {
    const float* x     = (const float*)d_in[0];
    const float* WQ    = (const float*)d_in[1];
    const float* bQ    = (const float*)d_in[2];
    const float* WK    = (const float*)d_in[3];
    const float* bK    = (const float*)d_in[4];
    const float* WV    = (const float*)d_in[5];
    const float* bV    = (const float*)d_in[6];
    const float* WO_w  = (const float*)d_in[7];
    const float* WO_b  = (const float*)d_in[8];
    const float* l1_w  = (const float*)d_in[9];
    const float* l1_b  = (const float*)d_in[10];
    const float* l2_w  = (const float*)d_in[11];
    const float* l2_b  = (const float*)d_in[12];
    const float* ln1_g = (const float*)d_in[13];
    const float* ln1_b = (const float*)d_in[14];
    const float* ln2_g = (const float*)d_in[15];
    const float* ln2_b = (const float*)d_in[16];
    (void)in_sizes; (void)n_in; (void)out_size; (void)ws_size;

    char* ws = (char*)d_ws;
    size_t off = 0;
    auto alloc = [&](size_t bytes) -> size_t {
        size_t o = off;
        off = (off + bytes + 255) & ~(size_t)255;
        return o;
    };

    const size_t oXH   = alloc((size_t)SEQ * DMODEL * 2);
    const size_t oWQKT = alloc((size_t)2 * DMODEL * DMODEL * 2);
    const size_t oWVT  = alloc((size_t)DMODEL * DMODEL * 2);
    const size_t oWOT  = alloc((size_t)DMODEL * DMODEL * 2);
    const size_t oL1T  = alloc((size_t)DMODEL * DFF * 2);
    const size_t oL2T  = alloc((size_t)DFF * DMODEL * 2);
    const size_t oBQK  = alloc((size_t)2 * DMODEL * 4);
    const size_t oQK   = alloc((size_t)SEQ * 2 * DMODEL * 2);
    const size_t oVT   = alloc((size_t)DMODEL * SEQ * 2);
    const size_t oATT  = alloc((size_t)SEQ * DMODEL * 2);
    const size_t oBLN1 = alloc((size_t)SEQ * DMODEL * 4);
    const size_t oSL1F = alloc((size_t)SEQ * DMODEL * 4);
    const size_t oSL1H = alloc((size_t)SEQ * DMODEL * 2);
    const size_t oFF1  = oQK;    // reuse QK+VT+ATT region (32MB)
    const size_t oA3   = oBLN1;  // reuse bln1

    half_t* XH   = (half_t*)(ws + oXH);
    half_t* WQKT = (half_t*)(ws + oWQKT);
    half_t* WVT  = (half_t*)(ws + oWVT);
    half_t* WOT  = (half_t*)(ws + oWOT);
    half_t* L1T  = (half_t*)(ws + oL1T);
    half_t* L2T  = (half_t*)(ws + oL2T);
    float*  BQK  = (float*)(ws + oBQK);
    half_t* QKh  = (half_t*)(ws + oQK);
    half_t* VTh  = (half_t*)(ws + oVT);
    half_t* ATTh = (half_t*)(ws + oATT);
    float*  BLN1 = (float*)(ws + oBLN1);
    float*  SL1F = (float*)(ws + oSL1F);
    half_t* SL1H = (half_t*)(ws + oSL1H);
    half_t* FF1H = (half_t*)(ws + oFF1);
    float*  A3F  = (float*)(ws + oA3);

    cvt_f32_f16<<<(SEQ * DMODEL) / 256, 256, 0, stream>>>(x, XH, SEQ * DMODEL);
    pack_qk_wt<<<(2 * DMODEL * DMODEL) / 256, 256, 0, stream>>>(WQ, WK, WQKT);
    pack_v_wt<<<(DMODEL * DMODEL) / 256, 256, 0, stream>>>(WV, WVT);
    cvt_t_f32_f16<<<(DMODEL * DMODEL) / 256, 256, 0, stream>>>(WO_w, WOT, DMODEL, DMODEL);
    cvt_t_f32_f16<<<(DMODEL * DFF) / 256, 256, 0, stream>>>(l1_w, L1T, DMODEL, DFF);
    cvt_t_f32_f16<<<(DFF * DMODEL) / 256, 256, 0, stream>>>(l2_w, L2T, DFF, DMODEL);
    concat2_f32<<<(2 * DMODEL) / 256, 256, 0, stream>>>(bQ, bK, BQK, DMODEL);

    // Q|K projection -> f16 [S,2048]
    gemm_wmma<<<dim3((2 * DMODEL) / GBN, SEQ / GBM), 256, 0, stream>>>(
        XH, WQKT, BQK, nullptr, QKh, nullptr, nullptr, SEQ, 2 * DMODEL, DMODEL, 0);

    // V projection, transposed output -> VT [1024][S] f16
    gemm_wmma<<<dim3(DMODEL / GBN, SEQ / GBM), 256, 0, stream>>>(
        XH, WVT, bV, nullptr, nullptr, nullptr, VTh, SEQ, DMODEL, DMODEL, 0);

    // multi-head attention
    attention_wmma<<<dim3(SEQ / 64, NHEAD), 128, 0, stream>>>(QKh, VTh, ATTh);

    // output projection + residual -> f32
    gemm_wmma<<<dim3(DMODEL / GBN, SEQ / GBM), 256, 0, stream>>>(
        ATTh, WOT, WO_b, x, nullptr, BLN1, nullptr, SEQ, DMODEL, DMODEL, 0);

    layernorm_row<<<SEQ, 256, 0, stream>>>(BLN1, ln1_g, ln1_b, SL1F, SL1H, DMODEL);

    // FFN up + ReLU -> f16
    gemm_wmma<<<dim3(DFF / GBN, SEQ / GBM), 256, 0, stream>>>(
        SL1H, L1T, l1_b, nullptr, FF1H, nullptr, nullptr, SEQ, DFF, DMODEL, 1);

    // FFN down + residual -> f32
    gemm_wmma<<<dim3(DMODEL / GBN, SEQ / GBM), 256, 0, stream>>>(
        FF1H, L2T, l2_b, SL1F, nullptr, A3F, nullptr, SEQ, DMODEL, DFF, 0);

    layernorm_row<<<SEQ, 256, 0, stream>>>(A3F, ln2_g, ln2_b, (float*)d_out, nullptr, DMODEL);
}
